// GINW_3layer_30339648979124
// MI455X (gfx1250) — matrix operand order repaired and checked by reading, not compiled, not run on hardware
//
#include <hip/hip_runtime.h>
#include <hip/hip_bf16.h>
#include <stdint.h>

#define N_NODES    10000
#define N_EDGES    640000
#define IN_CH      128
#define HID        256
#define OUT_CH     10
#define NUM_GRAPHS 64

typedef _Float16 v16h __attribute__((ext_vector_type(16)));
typedef float    v8f  __attribute__((ext_vector_type(8)));
typedef _Float16 h8   __attribute__((ext_vector_type(8)));

// ---------------- CSR build: histogram -> scan -> scatter ----------------

__global__ __launch_bounds__(256) void k_count(const int* __restrict__ ei,
                                               int* __restrict__ counts) {
  int e = blockIdx.x * blockDim.x + threadIdx.x;
  if (e < N_EDGES) atomicAdd(&counts[ei[N_EDGES + e]], 1);  // row 1 = dst
}

__global__ __launch_bounds__(1024) void k_scan(const int* __restrict__ counts,
                                               int* __restrict__ offs) {
  __shared__ int buf[1024];
  __shared__ int carry;
  const int tid = threadIdx.x;
  if (tid == 0) carry = 0;
  __syncthreads();
  for (int base = 0; base < N_NODES; base += 1024) {
    int i = base + tid;
    int v = (i < N_NODES) ? counts[i] : 0;
    buf[tid] = v;
    __syncthreads();
    for (int d = 1; d < 1024; d <<= 1) {
      int t = (tid >= d) ? buf[tid - d] : 0;
      __syncthreads();
      buf[tid] += t;
      __syncthreads();
    }
    int incl = buf[tid];
    int c    = carry;
    if (i < N_NODES) offs[i] = c + incl - v;   // exclusive prefix
    int tot = buf[1023];
    __syncthreads();
    if (tid == 0) carry = c + tot;
    __syncthreads();
  }
  if (tid == 0) offs[N_NODES] = carry;
}

__global__ __launch_bounds__(256) void k_scatter(const int* __restrict__ ei,
                                                 const float* __restrict__ ew,
                                                 const int* __restrict__ offs,
                                                 int* __restrict__ cursor,
                                                 int* __restrict__ csrc,
                                                 float* __restrict__ cw) {
  int e = blockIdx.x * blockDim.x + threadIdx.x;
  if (e < N_EDGES) {
    int d = ei[N_EDGES + e];
    int p = offs[d] + atomicAdd(&cursor[d], 1);
    csrc[p] = ei[e];        // row 0 = src
    cw[p]   = ew[e];
  }
}

// ---------------- fp32 -> fp16 weight conversion ----------------

__global__ __launch_bounds__(256) void k_cvt(const float* __restrict__ in,
                                             _Float16* __restrict__ out, int n) {
  int i = blockIdx.x * blockDim.x + threadIdx.x;
  if (i < n) out[i] = (_Float16)in[i];
}

// ---------------- GIN aggregation: h_pre[n] = x[n] + sum_e w_e * x[src_e] ----------------

template <int F, bool IN32>
__global__ __launch_bounds__(F) void k_conv(const float* __restrict__ xf,
                                            const _Float16* __restrict__ xh,
                                            const int* __restrict__ offs,
                                            const int* __restrict__ csrc,
                                            const float* __restrict__ cw,
                                            _Float16* __restrict__ out) {
  __shared__ int   s_s[F];
  __shared__ float s_w[F];
  const int n = blockIdx.x;
  const int f = threadIdx.x;
  float acc;
  if (IN32) acc = xf[(size_t)n * F + f];
  else      acc = (float)xh[(size_t)n * F + f];
  const int beg = offs[n], end = offs[n + 1];
  for (int base = beg; base < end; base += F) {
    int m = end - base;
    if (m > F) m = F;
    __syncthreads();
    if (f < m) { s_s[f] = csrc[base + f]; s_w[f] = cw[base + f]; }
    __syncthreads();
    for (int j = 0; j < m; ++j) {
      int   s = s_s[j];
      float w = s_w[j];
      if (j + 1 < m && (f & 31) == 0) {       // prefetch next gather row, 128B/lane-group
        int s2 = s_s[j + 1];
        if (IN32) __builtin_prefetch(xf + (size_t)s2 * F + f, 0, 1);
        else      __builtin_prefetch(xh + (size_t)s2 * F + f, 0, 1);
      }
      if (IN32) acc += w * xf[(size_t)s * F + f];
      else      acc += w * (float)xh[(size_t)s * F + f];
    }
  }
  out[(size_t)n * F + f] = (_Float16)acc;
}

// ---------------- WMMA GEMM: out = relu(A[M,K] @ W[N,K]^T + b), out fp16 [M,256] ----------------
// One wave per 16x16 output tile; M=10000 (625 tiles) x N=256 (16 tiles) = 10000 waves.

template <int K>
__global__ __launch_bounds__(256) void k_gemm(const _Float16* __restrict__ A,
                                              const _Float16* __restrict__ W,
                                              const float* __restrict__ bias,
                                              _Float16* __restrict__ out) {
  const int wave = threadIdx.x >> 5;
  const int lane = threadIdx.x & 31;
  const int tile = blockIdx.x * 8 + wave;      // 0..9999
  const int mt = tile >> 4;                    // 0..624
  const int nt = tile & 15;                    // 0..15
  const int m0 = mt * 16, n0 = nt * 16;
  const int r  = lane & 15;
  const int ha = (lane >> 4) * 8;              // A: K-halves interleaved by 8
  const int hb = (lane >> 4) * 8;              // B: K-halves interleaved by 8
  const _Float16* __restrict__ arow = A + (size_t)(m0 + r) * K;
  const _Float16* __restrict__ brow = W + (size_t)(n0 + r) * K;

  v8f c = {0.f, 0.f, 0.f, 0.f, 0.f, 0.f, 0.f, 0.f};
#pragma unroll
  for (int k0 = 0; k0 < K; k0 += 32) {
    h8 alo = *(const h8*)(arow + k0 + ha);        // VGPR0-3: K = k0+ha .. +7
    h8 ahi = *(const h8*)(arow + k0 + ha + 16);   // VGPR4-7: K = k0+16+ha .. +7
    h8 blo = *(const h8*)(brow + k0 + hb);        // VGPR0-3: K = k0+hb .. +7
    h8 bhi = *(const h8*)(brow + k0 + hb + 16);   // VGPR4-7: K = k0+16+hb .. +7
    v16h a, b;
#pragma unroll
    for (int i = 0; i < 8; ++i) {
      a[i] = alo[i]; a[i + 8] = ahi[i];
      b[i] = blo[i]; b[i + 8] = bhi[i];
    }
    c = __builtin_amdgcn_wmma_f32_16x16x32_f16(false, a, false, b, (short)0, c,
                                               false, false);
  }

  const int n  = n0 + r;
  const float bn = bias[n];
#pragma unroll
  for (int j = 0; j < 8; ++j) {
    float v = c[j] + bn;
    v = v > 0.f ? v : 0.f;
    out[(size_t)(m0 + ha + j) * HID + n] = (_Float16)v;  // row = m0 + lane_hi*8 + j
  }
}

// ---------------- pooling (batch is sorted -> contiguous graph ranges) ----------------

__global__ void k_bounds(const int* __restrict__ batch, int* __restrict__ gstart) {
  int g = blockIdx.x * blockDim.x + threadIdx.x;
  if (g > NUM_GRAPHS) return;
  if (g == NUM_GRAPHS) { gstart[g] = N_NODES; return; }
  int lo = 0, hi = N_NODES;
  while (lo < hi) {
    int mid = (lo + hi) >> 1;
    if (batch[mid] < g) lo = mid + 1; else hi = mid;
  }
  gstart[g] = lo;
}

__global__ __launch_bounds__(HID) void k_pool(const _Float16* __restrict__ h,
                                              const int* __restrict__ gstart,
                                              float* __restrict__ pooled) {
  const int g = blockIdx.x, f = threadIdx.x;
  const int beg = gstart[g], end = gstart[g + 1];
  float acc = 0.f;
  for (int n = beg; n < end; ++n) acc += (float)h[(size_t)n * HID + f];
  float cnt = (float)(end - beg);
  pooled[g * HID + f] = acc / fmaxf(cnt, 1.0f);
}

__global__ void k_final(const float* __restrict__ pooled,
                        const float* __restrict__ Wl,
                        const float* __restrict__ bl,
                        float* __restrict__ out) {
  int t = blockIdx.x * blockDim.x + threadIdx.x;
  if (t >= NUM_GRAPHS * OUT_CH) return;
  int g = t / OUT_CH, o = t % OUT_CH;
  float acc = bl[o];
  for (int k = 0; k < HID; ++k) acc += pooled[g * HID + k] * Wl[o * HID + k];
  out[t] = acc;
}

// ---------------- host launch ----------------

extern "C" void kernel_launch(void* const* d_in, const int* in_sizes, int n_in,
                              void* d_out, int out_size, void* d_ws, size_t ws_size,
                              hipStream_t stream) {
  const float* x    = (const float*)d_in[0];
  const int*   ei   = (const int*)d_in[1];
  const int*   batch= (const int*)d_in[2];
  const float* ew   = (const float*)d_in[3];
  const float* W1   = (const float*)d_in[4];
  const float* b1   = (const float*)d_in[5];
  const float* W2   = (const float*)d_in[6];
  const float* b2   = (const float*)d_in[7];
  const float* W3   = (const float*)d_in[8];
  const float* b3   = (const float*)d_in[9];
  const float* Wl   = (const float*)d_in[10];
  const float* bl   = (const float*)d_in[11];
  float* out = (float*)d_out;

  char* ws = (char*)d_ws;
  size_t off = 0;
  auto alloc = [&](size_t bytes) -> char* {
    char* p = ws + off;
    off = (off + bytes + 255) & ~(size_t)255;
    return p;
  };
  int*      counts = (int*)alloc((size_t)N_NODES * 4);
  int*      cursor = (int*)alloc((size_t)N_NODES * 4);
  int*      offs   = (int*)alloc((size_t)(N_NODES + 1) * 4);
  int*      gstart = (int*)alloc((size_t)(NUM_GRAPHS + 1) * 4);
  int*      csrc   = (int*)alloc((size_t)N_EDGES * 4);
  float*    cw     = (float*)alloc((size_t)N_EDGES * 4);
  _Float16* W1h    = (_Float16*)alloc((size_t)HID * IN_CH * 2);
  _Float16* W2h    = (_Float16*)alloc((size_t)HID * HID * 2);
  _Float16* W3h    = (_Float16*)alloc((size_t)HID * HID * 2);
  float*    pooled = (float*)alloc((size_t)NUM_GRAPHS * HID * 4);
  _Float16* bufP   = (_Float16*)alloc((size_t)N_NODES * HID * 2);
  _Float16* bufA   = (_Float16*)alloc((size_t)N_NODES * HID * 2);
  _Float16* bufB   = (_Float16*)alloc((size_t)N_NODES * HID * 2);

  hipMemsetAsync(counts, 0, (size_t)N_NODES * 4, stream);
  hipMemsetAsync(cursor, 0, (size_t)N_NODES * 4, stream);

  const int EB = (N_EDGES + 255) / 256;
  k_count  <<<EB, 256, 0, stream>>>(ei, counts);
  k_scan   <<<1, 1024, 0, stream>>>(counts, offs);
  k_scatter<<<EB, 256, 0, stream>>>(ei, ew, offs, cursor, csrc, cw);

  k_cvt<<<(HID * IN_CH + 255) / 256, 256, 0, stream>>>(W1, W1h, HID * IN_CH);
  k_cvt<<<(HID * HID   + 255) / 256, 256, 0, stream>>>(W2, W2h, HID * HID);
  k_cvt<<<(HID * HID   + 255) / 256, 256, 0, stream>>>(W3, W3h, HID * HID);

  const int GEMM_BLOCKS = (N_NODES / 16) * (HID / 16) / 8;  // 1250

  // layer 1
  k_conv<IN_CH, true ><<<N_NODES, IN_CH, 0, stream>>>(x, nullptr, offs, csrc, cw, bufP);
  k_gemm<IN_CH><<<GEMM_BLOCKS, 256, 0, stream>>>(bufP, W1h, b1, bufA);
  // layer 2
  k_conv<HID, false><<<N_NODES, HID, 0, stream>>>(nullptr, bufA, offs, csrc, cw, bufP);
  k_gemm<HID><<<GEMM_BLOCKS, 256, 0, stream>>>(bufP, W2h, b2, bufB);
  // layer 3
  k_conv<HID, false><<<N_NODES, HID, 0, stream>>>(nullptr, bufB, offs, csrc, cw, bufP);
  k_gemm<HID><<<GEMM_BLOCKS, 256, 0, stream>>>(bufP, W3h, b3, bufA);

  // mean pool + final linear
  k_bounds<<<1, 128, 0, stream>>>(batch, gstart);
  k_pool  <<<NUM_GRAPHS, HID, 0, stream>>>(bufA, gstart, pooled);
  k_final <<<(NUM_GRAPHS * OUT_CH + 127) / 128, 128, 0, stream>>>(pooled, Wl, bl, out);
}
